// MLPExperts_15848429322495
// MI455X (gfx1250) — matrix operand order repaired and checked by reading, not compiled, not run on hardware
//
#include <hip/hip_runtime.h>

// ---------------------------------------------------------------------------
// MLP experts: out = gelu(x @ wi) @ wo  per expert, bf16 WMMA compute path.
// E=8, T=2048 tokens/expert, H=1024, I=4096 (fp32 inputs/outputs).
// Pipeline: fp32->bf16 convert (x, wi, wo) -> GEMM1(+tanh-GELU, bf16 out)
//           -> GEMM2 (fp32 out).  Intermediate h (128MB bf16) rides in L2.
// GEMM: block tile 128x256xK32, 8 waves, each wave 64x64 (4x4 WMMA frags).
// A tile: global_load_async_to_lds_b128 (ASYNCcnt).  B tile: reg transpose.
// ---------------------------------------------------------------------------

typedef __attribute__((ext_vector_type(8)))  __bf16 bf16x8;
typedef __attribute__((ext_vector_type(16))) __bf16 bf16x16;
typedef __attribute__((ext_vector_type(8)))  float  v8f;

// ---------------- fp32 -> bf16 streaming conversion ------------------------
__global__ __launch_bounds__(256) void cvt_f32_bf16(const float* __restrict__ in,
                                                    __bf16* __restrict__ out,
                                                    long n) {
    long i = (((long)blockIdx.x * blockDim.x) + threadIdx.x) * 4;
    if (i + 3 < n) {
        float4 v = *reinterpret_cast<const float4*>(in + i);
        __bf16 o[4];
        o[0] = (__bf16)v.x; o[1] = (__bf16)v.y;
        o[2] = (__bf16)v.z; o[3] = (__bf16)v.w;
        *reinterpret_cast<uint2*>(out + i) = *reinterpret_cast<const uint2*>(o);
    } else {
        for (; i < n; ++i) out[i] = (__bf16)in[i];
    }
}

__device__ __forceinline__ float gelu_tanh(float x) {
    // branch-free GELU; tanh lowers to V_TANH_F32 on gfx1250
    const float u = 0.7978845608028654f * (x + 0.044715f * x * x * x);
    return 0.5f * x * (1.0f + tanhf(u));
}

// ---------------- tiled bf16 GEMM, 128x256x32, 8 waves ---------------------
// A: [M,K] row-major bf16, B: [K,N] row-major bf16 (stored transposed in LDS).
// GELU_OUT=1: C = bf16, gelu applied.  GELU_OUT=0: C = fp32 raw.
template <int GELU_OUT>
__global__ __launch_bounds__(256) void gemm_bf16_tile(
    const __bf16* __restrict__ Ab, const __bf16* __restrict__ Bb,
    void* __restrict__ Cb, int M, int N, int K,
    long batchA, long batchB, long batchC)
{
    constexpr int BM = 128, BN = 256, BK = 32;
    constexpr int LDA = 40;  // bf16 elems per LDS row (32 + 8 pad, 16B aligned)
    constexpr int LDB = 40;

    __shared__ __bf16 sA[BM * LDA];   // [m][k]      10 KB
    __shared__ __bf16 sB[BN * LDB];   // [n][k] (T)  20 KB

    const int e  = blockIdx.z;
    const __bf16* A = Ab + (size_t)e * batchA;
    const __bf16* B = Bb + (size_t)e * batchB;
    const int bm = blockIdx.x * BM;
    const int bn = blockIdx.y * BN;

    const int t    = threadIdx.x;
    const int lane = t & 31;
    const int wv   = t >> 5;
    const int wm   = wv >> 2;      // 0..1 : 64 rows per wave
    const int wn   = wv & 3;       // 0..3 : 64 cols per wave
    const int hl   = lane >> 4;    // lane half (WMMA VGPR layout)
    const int lr   = lane & 15;

    // global->LDS loader mapping
    const int arow = t >> 2;       // A rows: this row and +64
    const int agrp = t & 3;        // 4 groups of 8 bf16 across BK

    // LDS byte offsets for async A loads (low 32 bits of flat LDS address)
    const unsigned aLds0 = (unsigned)(size_t)&sA[arow * LDA + agrp * 8];
    const unsigned aLds1 = (unsigned)(size_t)&sA[(arow + 64) * LDA + agrp * 8];

    v8f acc[4][4];
    const v8f vzero = {0.f, 0.f, 0.f, 0.f, 0.f, 0.f, 0.f, 0.f};
#pragma unroll
    for (int fm = 0; fm < 4; ++fm)
#pragma unroll
        for (int fn = 0; fn < 4; ++fn) acc[fm][fn] = vzero;

    for (int kt = 0; kt < K; kt += BK) {
        // ---- stage B tile in registers (needs transpose) -------------------
        bf16x8 b0[2], b1[2];
#pragma unroll
        for (int j = 0; j < 2; ++j) {
            const int idx = t + j * 256;
            const int k2  = (idx >> 5) * 2;   // even k of (k,k+1) pair
            const int ng  = idx & 31;         // group of 8 columns
            b0[j] = *reinterpret_cast<const bf16x8*>(
                B + (size_t)(kt + k2) * N + bn + ng * 8);
            b1[j] = *reinterpret_cast<const bf16x8*>(
                B + (size_t)(kt + k2 + 1) * N + bn + ng * 8);
        }

        if (kt + BK < K) {  // hint next K-tile -> global_prefetch_b8
            __builtin_prefetch(A + (size_t)(bm + arow) * K + kt + BK, 0, 3);
            __builtin_prefetch(B + (size_t)(kt + BK + (t >> 5) * 2) * N + bn +
                                   (t & 31) * 8, 0, 3);
        }

        __syncthreads();  // previous tile fully consumed by all waves

        // ---- A tile: async copy global -> LDS (no VGPR round trip) ---------
        {
            const __bf16* ag0 = A + (size_t)(bm + arow) * K + kt + agrp * 8;
            const __bf16* ag1 = A + (size_t)(bm + arow + 64) * K + kt + agrp * 8;
            asm volatile("global_load_async_to_lds_b128 %0, %1, off"
                         :: "v"(aLds0), "v"(ag0) : "memory");
            asm volatile("global_load_async_to_lds_b128 %0, %1, off"
                         :: "v"(aLds1), "v"(ag1) : "memory");
        }

        // ---- B tile: pack (k,k+1) pairs, transposed b32 stores -------------
#pragma unroll
        for (int j = 0; j < 2; ++j) {
            const int idx = t + j * 256;
            const int k2  = (idx >> 5) * 2;
            const int ng  = idx & 31;
#pragma unroll
            for (int i = 0; i < 8; ++i) {
                union { __bf16 h[2]; unsigned u; } p;
                p.h[0] = b0[j][i]; p.h[1] = b1[j][i];
                *reinterpret_cast<unsigned*>(&sB[(ng * 8 + i) * LDB + k2]) = p.u;
            }
        }

        asm volatile("s_wait_asynccnt 0x0" ::: "memory");  // A tile landed
        __syncthreads();

        // ---- build fragments per ISA 16-bit A / B VGPR layouts -------------
        bf16x16 afr[4];
#pragma unroll
        for (int fm = 0; fm < 4; ++fm) {
            const int r = wm * 64 + fm * 16 + lr;
            bf16x8 lo = *reinterpret_cast<const bf16x8*>(&sA[r * LDA + (hl ? 8 : 0)]);
            bf16x8 hi = *reinterpret_cast<const bf16x8*>(&sA[r * LDA + (hl ? 24 : 16)]);
            afr[fm] = __builtin_shufflevector(lo, hi, 0, 1, 2, 3, 4, 5, 6, 7,
                                              8, 9, 10, 11, 12, 13, 14, 15);
        }
        bf16x16 bfr[4];
#pragma unroll
        for (int fn = 0; fn < 4; ++fn) {
            const int c = wn * 64 + fn * 16 + lr;
            bf16x8 lo = *reinterpret_cast<const bf16x8*>(&sB[c * LDB + hl * 8]);
            bf16x8 hi = *reinterpret_cast<const bf16x8*>(&sB[c * LDB + 16 + hl * 8]);
            bfr[fn] = __builtin_shufflevector(lo, hi, 0, 1, 2, 3, 4, 5, 6, 7,
                                              8, 9, 10, 11, 12, 13, 14, 15);
        }

#pragma unroll
        for (int fm = 0; fm < 4; ++fm)
#pragma unroll
            for (int fn = 0; fn < 4; ++fn)
                acc[fm][fn] = __builtin_amdgcn_wmma_f32_16x16x32_bf16(
                    /*neg_a=*/false, afr[fm], /*neg_b=*/false, bfr[fn],
                    /*c_mod=*/(short)0, acc[fm][fn],
                    /*reuse_a=*/false, /*reuse_b=*/false);
    }

    // epilogue: f32 C/D layout -> lane<16: M=v, lane>=16: M=8+v, N=lane%16
#pragma unroll
    for (int fm = 0; fm < 4; ++fm) {
        const int rowbase = bm + wm * 64 + fm * 16 + hl * 8;
#pragma unroll
        for (int fn = 0; fn < 4; ++fn) {
            const int col = bn + wn * 64 + fn * 16 + lr;
#pragma unroll
            for (int v = 0; v < 8; ++v) {
                const int r = rowbase + v;
                const float x = acc[fm][fn][v];
                if (GELU_OUT) {
                    reinterpret_cast<__bf16*>(Cb)[(size_t)e * batchC +
                                                  (size_t)r * N + col] =
                        (__bf16)gelu_tanh(x);
                } else {
                    reinterpret_cast<float*>(Cb)[(size_t)e * batchC +
                                                 (size_t)r * N + col] = x;
                }
            }
        }
    }
}

// ---------------------------------------------------------------------------
extern "C" void kernel_launch(void* const* d_in, const int* in_sizes, int n_in,
                              void* d_out, int out_size, void* d_ws, size_t ws_size,
                              hipStream_t stream) {
    const float* x  = (const float*)d_in[0];  // [1, 8, 2048, 1024]
    const float* wi = (const float*)d_in[1];  // [8, 1024, 4096]
    const float* wo = (const float*)d_in[2];  // [8, 4096, 1024]
    float* out = (float*)d_out;               // [1, 8, 2048, 1024]

    const int  E = 8, T = 2048, H = 1024, I = 4096;
    const long nx  = (long)E * T * H;   // 16,777,216
    const long nwi = (long)E * H * I;   // 33,554,432
    const long nwo = (long)E * I * H;   // 33,554,432

    char* ws = (char*)d_ws;               // 288 MB total layout:
    __bf16* xb  = (__bf16*)(ws);                              // 32 MB
    __bf16* wib = (__bf16*)(ws + nx * 2);                     // 64 MB
    __bf16* wob = (__bf16*)(ws + (nx + nwi) * 2);             // 64 MB
    __bf16* hb  = (__bf16*)(ws + (nx + nwi + nwo) * 2);       // 128 MB

    // fp32 -> bf16 (each thread converts 4 elements)
    cvt_f32_bf16<<<(unsigned)(nx  / 1024), 256, 0, stream>>>(x,  xb,  nx);
    cvt_f32_bf16<<<(unsigned)(nwi / 1024), 256, 0, stream>>>(wi, wib, nwi);
    cvt_f32_bf16<<<(unsigned)(nwo / 1024), 256, 0, stream>>>(wo, wob, nwo);

    // GEMM1: h = gelu(x @ wi)   [2048x1024]x[1024x4096] per expert
    gemm_bf16_tile<1><<<dim3(T / 128, I / 256, E), 256, 0, stream>>>(
        xb, wib, hb, T, I, H, (long)T * H, (long)H * I, (long)T * I);

    // GEMM2: out = h @ wo       [2048x4096]x[4096x1024] per expert
    gemm_bf16_tile<0><<<dim3(T / 128, H / 256, E), 256, 0, stream>>>(
        hb, wob, out, T, H, I, (long)T * I, (long)I * H, (long)T * H);
}
